// scnn_conv_670014898740
// MI455X (gfx1250) — matrix operand order repaired and checked by reading, not compiled, not run on hardware
//
#include <hip/hip_runtime.h>

// Spectral ChebNet conv for MI455X (gfx1250, wave32).
//   y = einsum('nik,iok->no', [x, T1..T3(Ll), T1..T3(Lu)], W)
// L converted to f16 (134MB < 192MB L2 -> Chebyshev passes 2/3 hit L2), six
// dependent GEMMs via v_wmma_f32_16x16x32_f16 (f32 accumulate), recurrence
// (2*L*T_cur - T_prev) fused into the split-K reduction epilogue, tiny fp32
// combine with W at the end.
// Per-wave GEMM tile: 32x32 (2 A frags x 2 B frags -> 4 WMMA / K=32 step).
// Split-K partials stored in C-fragment-native layout => all-b128 epilogue.
// Workspace: 128MiB L16 + 16MiB partials + 7*512KiB terms (~148MiB).

#define M_DIM 8192
#define K_DIM 8192
#define F 32
#define SPLITK 16
#define NT ((M_DIM / 16) * 2)   // 1024 16x16 C tiles (512 m-tiles x 2 n-halves)

typedef __attribute__((ext_vector_type(16))) _Float16 v16h;
typedef __attribute__((ext_vector_type(8)))  _Float16 h8;
typedef __attribute__((ext_vector_type(8)))  float    v8f;
typedef __attribute__((ext_vector_type(4)))  float    f4;

union AFrag { v16h v; h8 h[2]; };

// ---------------- fp32 -> f16 elementwise (8 elems/thread) -------------------
__global__ __launch_bounds__(256)
void k_cvt_f16(const float* __restrict__ in, _Float16* __restrict__ out) {
    size_t base = ((size_t)blockIdx.x * 256 + threadIdx.x) * 8;
    f4 a = __builtin_nontemporal_load((const f4*)(in + base));      // NT: keep fp32 L out of L2
    f4 b = __builtin_nontemporal_load(((const f4*)(in + base)) + 1);
    h8 o;
    o[0] = (_Float16)a[0]; o[1] = (_Float16)a[1];
    o[2] = (_Float16)a[2]; o[3] = (_Float16)a[3];
    o[4] = (_Float16)b[0]; o[5] = (_Float16)b[1];
    o[6] = (_Float16)b[2]; o[7] = (_Float16)b[3];
    *(h8*)(out + base) = o;   // f16 L stays resident in L2 for passes 2,3
}

// ------------- x [M,32] fp32 -> xT [32,M] f16 (transposed operand) -----------
__global__ __launch_bounds__(256)
void k_xT(const float* __restrict__ x, _Float16* __restrict__ xT) {
    int tid = blockIdx.x * 256 + threadIdx.x;   // 0 .. M*F-1
    int i = tid >> 13;                          // feature 0..31
    int n = tid & (M_DIM - 1);                  // row 0..8191
    xT[(size_t)i * M_DIM + n] = (_Float16)x[(size_t)n * F + i];
}

// ------------------- split-K WMMA GEMM: P[s] = A_chunk @ B ------------------
// A  : [8192,8192] f16 row-major (the f16 Laplacian).
// Bt : [32,8192] f16 transposed operand (Bt[n][k] = B[k][n]).
// P  : [SPLITK][NT][32 lanes][8] fp32 partials, C-fragment-native layout.
// Per wave: 32 rows x 32 cols = 4 16x16 tiles, 4 WMMA per K=32 step.
// A frag (ISA 7.12.2, 16-bit A 16x32): lane m=lane&15; VGPR0-3 K=kh*8+0..7,
// VGPR4-7 K=16+kh*8+0..7 -> two contiguous B128 loads per lane.
// B frag (32x16): lane n=lane&15; halves 0-7 K=kh*8+0..7, halves 8-15
// K=16+kh*8+0..7 -> two contiguous B128 loads of row Bt[n] (transposed layout).
__global__ __launch_bounds__(256)
void k_gemm(const _Float16* __restrict__ A,
            const _Float16* __restrict__ Bt,
            float* __restrict__ P) {
    const int lane = threadIdx.x & 31;
    const int wave = threadIdx.x >> 5;
    const int r16  = lane & 15;
    const int kh   = lane >> 4;
    const int m0   = blockIdx.x * 256 + wave * 32;        // 32 rows per wave
    const int kbeg = blockIdx.y * (K_DIM / SPLITK);
    const int kend = kbeg + (K_DIM / SPLITK);

    const _Float16* arow0 = A  + (size_t)(m0 + r16)      * K_DIM + kh * 8;
    const _Float16* arow1 = A  + (size_t)(m0 + 16 + r16) * K_DIM + kh * 8;
    const _Float16* b0r   = Bt + (size_t)(r16)           * K_DIM + kh * 8;
    const _Float16* b1r   = Bt + (size_t)(r16 + 16)      * K_DIM + kh * 8;

    v8f acc00 = {}, acc01 = {}, acc10 = {}, acc11 = {};

#pragma unroll 2
    for (int k = kbeg; k < kend; k += 32) {
        AFrag a0, a1;
        a0.h[0] = *(const h8*)(arow0 + k);
        a0.h[1] = *(const h8*)(arow0 + k + 16);
        a1.h[0] = *(const h8*)(arow1 + k);
        a1.h[1] = *(const h8*)(arow1 + k + 16);
        AFrag bf0, bf1;
        bf0.h[0] = *(const h8*)(b0r + k);
        bf0.h[1] = *(const h8*)(b0r + k + 16);
        bf1.h[0] = *(const h8*)(b1r + k);
        bf1.h[1] = *(const h8*)(b1r + k + 16);
        v16h b0 = bf0.v;
        v16h b1 = bf1.v;
        acc00 = __builtin_amdgcn_wmma_f32_16x16x32_f16(
                    false, a0.v, false, b0, (short)0, acc00, false, false);
        acc01 = __builtin_amdgcn_wmma_f32_16x16x32_f16(
                    false, a0.v, false, b1, (short)0, acc01, false, false);
        acc10 = __builtin_amdgcn_wmma_f32_16x16x32_f16(
                    false, a1.v, false, b0, (short)0, acc10, false, false);
        acc11 = __builtin_amdgcn_wmma_f32_16x16x32_f16(
                    false, a1.v, false, b1, (short)0, acc11, false, false);
    }

    // C-fragment-native store: tile = tm*2 + bn; lane-contiguous v8f (2xb128).
    const int tb = (m0 >> 4) * 2;   // first of this wave's 4 tiles
    float* pw = P + (((size_t)blockIdx.y * NT + tb) * 32 + lane) * 8;
    *(v8f*)(pw)       = acc00;      // (tm0, n0)
    *(v8f*)(pw + 256) = acc01;      // (tm0, n1)
    *(v8f*)(pw + 512) = acc10;      // (tm0+1, n0)
    *(v8f*)(pw + 768) = acc11;      // (tm0+1, n1)
}

// ------- reduce split-K partials + Chebyshev epilogue, emit f16 transposed ----
// Ct[n][m] = f16( alpha * sum_s P[s][tile][lane][r] + beta * Cprev[n][m] )
__global__ __launch_bounds__(256)
void k_reduce(const float* __restrict__ P,
              const _Float16* __restrict__ Cprev,
              float alpha, float beta,
              _Float16* __restrict__ Ct) {
    int tid  = blockIdx.x * 256 + threadIdx.x;   // 0 .. NT*32-1
    int lane = tid & 31;
    int tile = tid >> 5;
    const float* p = P + ((size_t)tile * 32 + lane) * 8;
    v8f s = {};
#pragma unroll
    for (int si = 0; si < SPLITK; ++si)
        s += *(const v8f*)(p + (size_t)si * NT * 32 * 8);

    // decode fragment coords: element r -> M = tm*16 + 8*(lane>>4) + r, N = bn*16 + (lane&15)
    int tm = tile >> 1, bn = tile & 1;
    int mb = tm * 16 + 8 * (lane >> 4);
    int n  = bn * 16 + (lane & 15);
    h8 cp = *(const h8*)(Cprev + (size_t)n * M_DIM + mb);
    h8 o;
#pragma unroll
    for (int r = 0; r < 8; ++r)
        o[r] = (_Float16)(alpha * s[r] + beta * (float)cp[r]);
    *(h8*)(Ct + (size_t)n * M_DIM + mb) = o;
}

// ------------------- y[n,o] = sum_i sum_k X[n,i,k] * W[i,o,k] ----------------
__global__ __launch_bounds__(256)
void k_combine(const float* __restrict__ x,
               const _Float16* __restrict__ t0, const _Float16* __restrict__ t1,
               const _Float16* __restrict__ t2, const _Float16* __restrict__ t3,
               const _Float16* __restrict__ t4, const _Float16* __restrict__ t5,
               const float* __restrict__ W, float* __restrict__ y) {
    __shared__ float Ws[F * F * 7];
    for (int j = threadIdx.x; j < F * F * 7; j += 256) Ws[j] = W[j];
    __syncthreads();
    int n = blockIdx.x * 8 + (threadIdx.x >> 5);
    int o = threadIdx.x & 31;
    float acc = 0.f;
#pragma unroll
    for (int i = 0; i < F; ++i) {
        const float* w = &Ws[(i * F + o) * 7];     // stride 7 mod 64 -> no bank conflict
        acc += x[(size_t)n * F + i]             * w[0];
        acc += (float)t0[(size_t)i * M_DIM + n] * w[1];
        acc += (float)t1[(size_t)i * M_DIM + n] * w[2];
        acc += (float)t2[(size_t)i * M_DIM + n] * w[3];
        acc += (float)t3[(size_t)i * M_DIM + n] * w[4];
        acc += (float)t4[(size_t)i * M_DIM + n] * w[5];
        acc += (float)t5[(size_t)i * M_DIM + n] * w[6];
    }
    y[(size_t)n * F + o] = acc;
}

extern "C" void kernel_launch(void* const* d_in, const int* in_sizes, int n_in,
                              void* d_out, int out_size, void* d_ws, size_t ws_size,
                              hipStream_t stream) {
    const float* x  = (const float*)d_in[0];   // [8192,32]
    const float* Ll = (const float*)d_in[1];   // [8192,8192]
    const float* Lu = (const float*)d_in[2];   // [8192,8192]
    const float* W  = (const float*)d_in[3];   // [32,32,7]
    float* y = (float*)d_out;                  // [8192,32]

    char* ws = (char*)d_ws;
    _Float16* L16 = (_Float16*)ws;                                   // 128 MiB (reused Ll->Lu)
    float*    P   = (float*)(ws + (size_t)M_DIM * K_DIM * 2);        // 16 MiB split-K partials
    _Float16* xT  = (_Float16*)((char*)P + (size_t)SPLITK * NT * 32 * 8 * 4);
    _Float16* T[6];
    for (int j = 0; j < 6; ++j) T[j] = xT + (size_t)(j + 1) * F * M_DIM;

    const dim3 blk(256);
    const dim3 ggrid(M_DIM / 256, SPLITK);               // 512 blocks, 4096 waves
    const int  cvtBlocks = (int)(((size_t)M_DIM * K_DIM) / (256 * 8));  // 32768
    const int  mfBlocks  = (M_DIM * F) / 256;            // 1024
    const int  rdBlocks  = (NT * 32) / 256;              // 128

    // x -> transposed f16 operand
    k_xT<<<mfBlocks, blk, 0, stream>>>(x, xT);

    // ---- Ll phase: T1=Ll x ; T2=2 Ll T1 - x ; T3=2 Ll T2 - T1 ----
    k_cvt_f16<<<cvtBlocks, blk, 0, stream>>>(Ll, L16);
    k_gemm<<<ggrid, blk, 0, stream>>>(L16, xT, P);
    k_reduce<<<rdBlocks, blk, 0, stream>>>(P, xT, 1.f, 0.f, T[0]);
    k_gemm<<<ggrid, blk, 0, stream>>>(L16, T[0], P);
    k_reduce<<<rdBlocks, blk, 0, stream>>>(P, xT, 2.f, -1.f, T[1]);
    k_gemm<<<ggrid, blk, 0, stream>>>(L16, T[1], P);
    k_reduce<<<rdBlocks, blk, 0, stream>>>(P, T[0], 2.f, -1.f, T[2]);

    // ---- Lu phase ----
    k_cvt_f16<<<cvtBlocks, blk, 0, stream>>>(Lu, L16);
    k_gemm<<<ggrid, blk, 0, stream>>>(L16, xT, P);
    k_reduce<<<rdBlocks, blk, 0, stream>>>(P, xT, 1.f, 0.f, T[3]);
    k_gemm<<<ggrid, blk, 0, stream>>>(L16, T[3], P);
    k_reduce<<<rdBlocks, blk, 0, stream>>>(P, xT, 2.f, -1.f, T[4]);
    k_gemm<<<ggrid, blk, 0, stream>>>(L16, T[4], P);
    k_reduce<<<rdBlocks, blk, 0, stream>>>(P, T[3], 2.f, -1.f, T[5]);

    // ---- final 7-term contraction with W ----
    k_combine<<<M_DIM / 8, blk, 0, stream>>>(x, T[0], T[1], T[2], T[3], T[4], T[5], W, y);
}
